// AttentionBlock_11879879541874
// MI455X (gfx1250) — matrix operand order repaired and checked by reading, not compiled, not run on hardware
//
#include <hip/hip_runtime.h>
#include <hip/hip_bf16.h>
#include <math.h>

// ---------------------------------------------------------------------------
// AttentionBlock for MI455X (gfx1250, wave32, WMMA 16x16x32 f16)
//   B=2, H=W=64 -> N=4096 tokens, C=512, GROUPS=32
// Compute-bound (~86 GFLOP vs ~40MB traffic) -> all GEMMs on v_wmma_f32_16x16x32_f16.
// Attention fused (flash-style, online softmax) to avoid 128MB S matrix.
// K tiles staged via GLOBAL_LOAD_ASYNC_TO_LDS_B128 (ASYNCcnt), double-buffered.
// ---------------------------------------------------------------------------

#define NTOK 4096
#define CCH  512
#define NGRP 32
#define BATCH 2

typedef _Float16 v16h __attribute__((ext_vector_type(16)));
typedef _Float16 v8h  __attribute__((ext_vector_type(8)));
typedef _Float16 v4h  __attribute__((ext_vector_type(4)));
typedef float    v8f  __attribute__((ext_vector_type(8)));
typedef int      v4i  __attribute__((vector_size(16)));

static __device__ __forceinline__ v8h ld8(const _Float16* p) {
    return *(const v8h*)p;
}
static __device__ __forceinline__ v16h mk16(v8h lo, v8h hi) {
    v16h r;
#pragma unroll
    for (int i = 0; i < 8; ++i) { r[i] = lo[i]; r[i + 8] = hi[i]; }
    return r;
}
static __device__ __forceinline__ v8f wmma16(v16h a, v16h b, v8f c) {
    // D = A(16x32 f16) * B(32x16 f16) + C(16x16 f32)
    return __builtin_amdgcn_wmma_f32_16x16x32_f16(false, a, false, b,
                                                  (short)0, c, false, false);
}

// --- CDNA5 async global->LDS copy (16B per lane), ASYNCcnt tracked ---------
#if __has_builtin(__builtin_amdgcn_global_load_async_to_lds_b128)
static __device__ __forceinline__ void async_cp16(const _Float16* g, _Float16* l) {
    __builtin_amdgcn_global_load_async_to_lds_b128(
        (__attribute__((address_space(1))) v4i*)(g),
        (__attribute__((address_space(3))) v4i*)(l), 0, 0);
}
#else
static __device__ __forceinline__ void async_cp16(const _Float16* g, _Float16* l) {
    unsigned loff = (unsigned)(unsigned long long)
        (__attribute__((address_space(3))) _Float16*)l;
    unsigned long long ga = (unsigned long long)g;
    asm volatile("global_load_async_to_lds_b128 %0, %1, off"
                 :: "v"(loff), "v"(ga) : "memory");
}
#endif

#if __has_builtin(__builtin_amdgcn_s_wait_asynccnt)
#define WAIT_ASYNC(n) __builtin_amdgcn_s_wait_asynccnt(n)
#else
#define WAIT_ASYNC(n) asm volatile("s_wait_asynccnt %0" :: "i"(n) : "memory")
#endif

// ---------------------------------------------------------------------------
// 1) GroupNorm statistics: one block per (b, group); reduce 64*64*16 elements.
// ---------------------------------------------------------------------------
__global__ __launch_bounds__(256) void gn_stats_kernel(const float* __restrict__ x,
                                                       float* __restrict__ stats) {
    const int bg = blockIdx.x;           // 0..63
    const int b = bg >> 5, g = bg & 31;
    const int t = threadIdx.x;
    float s = 0.f, q = 0.f;
    // 4096 pixels * 16 channels = 65536 elements per (b,g)
    for (int f = t; f < 65536; f += 256) {
        const int p = f >> 4, c = f & 15;
        const float v = x[((size_t)b * NTOK + p) * CCH + g * 16 + c];
        s += v; q += v * v;
    }
    __shared__ float rs[256], rq[256];
    rs[t] = s; rq[t] = q;
    __syncthreads();
    for (int off = 128; off > 0; off >>= 1) {
        if (t < off) { rs[t] += rs[t + off]; rq[t] += rq[t + off]; }
        __syncthreads();
    }
    if (t == 0) {
        const float mean = rs[0] * (1.f / 65536.f);
        const float var  = rq[0] * (1.f / 65536.f) - mean * mean;
        stats[2 * bg]     = mean;
        stats[2 * bg + 1] = rsqrtf(var + 1e-5f);
    }
}

// ---------------------------------------------------------------------------
// 2) GroupNorm apply + convert to f16 (h_norm feeds all WMMA GEMMs).
// ---------------------------------------------------------------------------
__global__ __launch_bounds__(256) void gn_apply_kernel(const float* __restrict__ x,
                                                       const float* __restrict__ gamma,
                                                       const float* __restrict__ beta,
                                                       const float* __restrict__ stats,
                                                       _Float16* __restrict__ hN) {
    const size_t i4 = (size_t)blockIdx.x * 256 + threadIdx.x;   // 1M threads
    const size_t base = i4 * 4;
    const int c = (int)(base & 511);
    const int b = (int)(base >> 21);          // / (4096*512)
    const int g = c >> 4;
    const float mean = stats[2 * ((b << 5) + g)];
    const float rstd = stats[2 * ((b << 5) + g) + 1];
    const float4 xv = *(const float4*)(x + base);
    const float4 gv = *(const float4*)(gamma + c);
    const float4 bv = *(const float4*)(beta + c);
    v4h o;
    o[0] = (_Float16)((xv.x - mean) * rstd * gv.x + bv.x);
    o[1] = (_Float16)((xv.y - mean) * rstd * gv.y + bv.y);
    o[2] = (_Float16)((xv.z - mean) * rstd * gv.z + bv.z);
    o[3] = (_Float16)((xv.w - mean) * rstd * gv.w + bv.w);
    *(v4h*)(hN + base) = o;
}

// ---------------------------------------------------------------------------
// 3) Weight pack: WT[d][c] = (f16) W[c][d]  (so B-fragments are contiguous).
// ---------------------------------------------------------------------------
__global__ __launch_bounds__(256) void pack_w_kernel(const float* __restrict__ w,
                                                     _Float16* __restrict__ wt) {
    const int idx = blockIdx.x * 256 + threadIdx.x;   // 262144
    const int c = idx & 511, d = idx >> 9;
    wt[(size_t)d * CCH + c] = (_Float16)w[(size_t)c * CCH + d];
}

// ---------------------------------------------------------------------------
// 4) WMMA GEMM: Y = A(4096x512 f16) * W(512x512) + bias -> f16.
//    8 waves/block, each wave 16 rows x 64 cols (4 wmma col-tiles, 16 K-steps).
//    trans==1 stores Y transposed ([d][n]) -> used for V so PV B-frags are
//    contiguous in the flash kernel.
// ---------------------------------------------------------------------------
__global__ __launch_bounds__(256) void qkv_gemm_kernel(const _Float16* __restrict__ A,
                                                       const _Float16* __restrict__ wt,
                                                       const float* __restrict__ bias,
                                                       _Float16* __restrict__ out,
                                                       int trans) {
    const int tid = threadIdx.x;
    const int wave = tid >> 5, lane = tid & 31;
    const int hal = lane >> 4, n = lane & 15;
    const int rows0 = blockIdx.x * 128 + wave * 16;
    const int col0  = blockIdx.y * 64;
    const _Float16* Ab = A + (size_t)blockIdx.z * NTOK * CCH;
    _Float16* outb = out + (size_t)blockIdx.z * NTOK * CCH;

    const _Float16* arow = Ab + (size_t)(rows0 + n) * CCH;
    v8f acc[4] = {};
#pragma unroll 4
    for (int kc = 0; kc < 16; ++kc) {
        const int kb = kc * 32;
        const v16h a = mk16(ld8(arow + kb + 16 * hal), ld8(arow + kb + 16 * hal + 8));
#pragma unroll
        for (int j = 0; j < 4; ++j) {
            const _Float16* brow = wt + (size_t)(col0 + j * 16 + n) * CCH + kb + 16 * hal;
            acc[j] = wmma16(a, mk16(ld8(brow), ld8(brow + 8)), acc[j]);
        }
    }
#pragma unroll
    for (int j = 0; j < 4; ++j) {
        const int d = col0 + j * 16 + n;
        const float bj = bias[d];
#pragma unroll
        for (int r = 0; r < 8; ++r) {
            const int row = rows0 + r + 8 * hal;
            const float v = acc[j][r] + bj;
            if (trans)
                outb[(size_t)d * NTOK + row] = (_Float16)v;   // Vt[c][token]
            else
                outb[(size_t)row * CCH + d] = (_Float16)v;    // row-major
        }
    }
}

// ---------------------------------------------------------------------------
// 5) Flash attention: O = softmax(Q K^T / sqrt(C)) V, fused, online softmax.
//    Block = 8 waves: 4 row-groups (16 queries each) x 2 col-groups (256 ch).
//    K tiles (32 keys x 512) double-buffered in padded LDS via async copies:
//    tile kt+1 streams in (ASYNCcnt) while tile kt feeds 48 WMMAs + softmax.
//    P re-layout (accum -> A-frag) goes through a per-wave LDS buffer.
// ---------------------------------------------------------------------------
#define KPAD 520   // 32x520 f16, 1040B row stride (16B aligned, bank-friendly)

static __device__ __forceinline__ void stage_ktile(const _Float16* __restrict__ Kb,
                                                   _Float16* kt_buf, int kt, int tid) {
#pragma unroll
    for (int i = 0; i < 8; ++i) {
        const int cid = tid + i * 256;
        const int row = cid >> 6, c8 = (cid & 63) * 8;
        async_cp16(&Kb[(size_t)(kt * 32 + row) * CCH + c8], &kt_buf[row * KPAD + c8]);
    }
}

__global__ __launch_bounds__(256) void flash_attn_kernel(const _Float16* __restrict__ Q,
                                                         const _Float16* __restrict__ K,
                                                         const _Float16* __restrict__ Vt,
                                                         _Float16* __restrict__ O) {
    __shared__ _Float16 Kt[2][32 * KPAD];     // double-buffered K tile (66.5 KB)
    __shared__ _Float16 Pb[8][512];           // per-wave 16x32 P tile (8 KB)

    const int tid = threadIdx.x;
    const int wave = tid >> 5, lane = tid & 31;
    const int hal = lane >> 4, n = lane & 15;
    const int rowg = wave >> 1, colg = wave & 1;
    const int b = blockIdx.y;
    const int qrow0 = blockIdx.x * 64 + rowg * 16;

    const _Float16* Qb = Q + (size_t)b * NTOK * CCH;
    const _Float16* Kb = K + (size_t)b * NTOK * CCH;
    const _Float16* Vb = Vt + (size_t)b * NTOK * CCH;  // [C][N]
    const _Float16* qrow = Qb + (size_t)(qrow0 + n) * CCH;
    _Float16* P = Pb[wave];

    v8f o[16] = {};                 // 16 rows x 256 cols f32 accum (128 VGPRs)
    float rmax[8], rsum[8];
#pragma unroll
    for (int r = 0; r < 8; ++r) { rmax[r] = -3.0e38f; rsum[r] = 0.f; }
    const float scale = 0.04419417382415922f;  // 512^-0.5

    // prologue: stage tile 0 into buffer 0
    stage_ktile(Kb, Kt[0], 0, tid);

    for (int kt = 0; kt < NTOK / 32; ++kt) {
        const int cur = kt & 1;
        // overlap: issue next tile's async copies into the other buffer
        if (kt + 1 < NTOK / 32) {
            stage_ktile(Kb, Kt[cur ^ 1], kt + 1, tid);
            WAIT_ASYNC(8);   // 16 outstanding -> wait until tile kt complete
        } else {
            WAIT_ASYNC(0);
        }
        __syncthreads();     // cross-wave visibility of async LDS writes
        const _Float16* Ktc = Kt[cur];

        // S tile: 16 queries x 32 keys, K-dim = 512
        v8f s0 = {}, s1 = {};
#pragma unroll 4
        for (int kc = 0; kc < 16; ++kc) {
            const int kb = kc * 32;
            const v16h a = mk16(ld8(qrow + kb + 16 * hal),
                                ld8(qrow + kb + 16 * hal + 8));
            const _Float16* k0 = &Ktc[n * KPAD + kb + 16 * hal];
            const _Float16* k1 = &Ktc[(16 + n) * KPAD + kb + 16 * hal];
            s0 = wmma16(a, mk16(ld8(k0), ld8(k0 + 8)), s0);
            s1 = wmma16(a, mk16(ld8(k1), ld8(k1 + 8)), s1);
        }

        // online softmax (rows live in 16-lane groups of one VGPR)
#pragma unroll
        for (int r = 0; r < 8; ++r) {
            const float a0 = s0[r] * scale, a1 = s1[r] * scale;
            float m = fmaxf(a0, a1);
            m = fmaxf(m, __shfl_xor(m, 1));
            m = fmaxf(m, __shfl_xor(m, 2));
            m = fmaxf(m, __shfl_xor(m, 4));
            m = fmaxf(m, __shfl_xor(m, 8));
            const float nm = fmaxf(rmax[r], m);
            const float corr = __expf(rmax[r] - nm);
            rmax[r] = nm;
            const float p0 = __expf(a0 - nm), p1 = __expf(a1 - nm);
            float ts = p0 + p1;
            ts += __shfl_xor(ts, 1);
            ts += __shfl_xor(ts, 2);
            ts += __shfl_xor(ts, 4);
            ts += __shfl_xor(ts, 8);
            rsum[r] = rsum[r] * corr + ts;
#pragma unroll
            for (int j = 0; j < 16; ++j) o[j][r] *= corr;
            const int row = r + 8 * hal;
            P[row * 32 + n]      = (_Float16)p0;
            P[row * 32 + 16 + n] = (_Float16)p1;
        }

        // PV: A = P (16x32), B = V chunk (32 keys x 16 ch) from Vt (contiguous)
        const v16h aP = mk16(ld8(&P[n * 32 + 16 * hal]),
                             ld8(&P[n * 32 + 16 * hal + 8]));
#pragma unroll
        for (int j = 0; j < 16; ++j) {
            const int cc = colg * 256 + j * 16 + n;
            const _Float16* vrow = &Vb[(size_t)cc * NTOK + kt * 32 + 16 * hal];
            o[j] = wmma16(aP, mk16(ld8(vrow), ld8(vrow + 8)), o[j]);
        }
        __syncthreads();     // all waves done reading Kt[cur] before overwrite
    }

    _Float16* Ob = O + (size_t)b * NTOK * CCH;
#pragma unroll
    for (int r = 0; r < 8; ++r) {
        const float inv = 1.f / rsum[r];
        const int row = qrow0 + r + 8 * hal;
#pragma unroll
        for (int j = 0; j < 16; ++j) {
            const int d = colg * 256 + j * 16 + n;
            Ob[(size_t)row * CCH + d] = (_Float16)(o[j][r] * inv);
        }
    }
}

// ---------------------------------------------------------------------------
// 6) Output projection + bias + residual, f32 out.
// ---------------------------------------------------------------------------
__global__ __launch_bounds__(256) void proj_gemm_kernel(const _Float16* __restrict__ A,
                                                        const _Float16* __restrict__ wt,
                                                        const float* __restrict__ bias,
                                                        const float* __restrict__ x,
                                                        float* __restrict__ out) {
    const int tid = threadIdx.x;
    const int wave = tid >> 5, lane = tid & 31;
    const int hal = lane >> 4, n = lane & 15;
    const int rows0 = blockIdx.x * 128 + wave * 16;
    const int col0  = blockIdx.y * 64;
    const size_t boff = (size_t)blockIdx.z * NTOK * CCH;
    const _Float16* Ab = A + boff;

    const _Float16* arow = Ab + (size_t)(rows0 + n) * CCH;
    v8f acc[4] = {};
#pragma unroll 4
    for (int kc = 0; kc < 16; ++kc) {
        const int kb = kc * 32;
        const v16h a = mk16(ld8(arow + kb + 16 * hal), ld8(arow + kb + 16 * hal + 8));
#pragma unroll
        for (int j = 0; j < 4; ++j) {
            const _Float16* brow = wt + (size_t)(col0 + j * 16 + n) * CCH + kb + 16 * hal;
            acc[j] = wmma16(a, mk16(ld8(brow), ld8(brow + 8)), acc[j]);
        }
    }
#pragma unroll
    for (int j = 0; j < 4; ++j) {
        const int d = col0 + j * 16 + n;
        const float bj = bias[d];
#pragma unroll
        for (int r = 0; r < 8; ++r) {
            const int row = rows0 + r + 8 * hal;
            const size_t gi = boff + (size_t)row * CCH + d;
            out[gi] = acc[j][r] + bj + x[gi];
        }
    }
}

// ---------------------------------------------------------------------------
// Launch: GN stats -> GN apply -> pack weights -> QKV GEMMs -> flash -> proj.
// Workspace layout (~42 MB total).
// ---------------------------------------------------------------------------
extern "C" void kernel_launch(void* const* d_in, const int* in_sizes, int n_in,
                              void* d_out, int out_size, void* d_ws, size_t ws_size,
                              hipStream_t stream) {
    const float* x     = (const float*)d_in[0];
    const float* gamma = (const float*)d_in[1];
    const float* beta  = (const float*)d_in[2];
    const float* wq    = (const float*)d_in[3];
    const float* bq    = (const float*)d_in[4];
    const float* wk    = (const float*)d_in[5];
    const float* bk    = (const float*)d_in[6];
    const float* wv    = (const float*)d_in[7];
    const float* bv    = (const float*)d_in[8];
    const float* wp    = (const float*)d_in[9];
    const float* bp    = (const float*)d_in[10];
    float* out = (float*)d_out;

    const size_t NC   = (size_t)NTOK * CCH;                 // 2M elems
    const size_t hBytes = (size_t)BATCH * NC * sizeof(_Float16);  // 8 MB
    const size_t wBytes = (size_t)CCH * CCH * sizeof(_Float16);   // 0.5 MB

    uint8_t* w = (uint8_t*)d_ws;
    size_t off = 0;
    float* stats = (float*)(w + off);           off += 1024;
    _Float16* hN  = (_Float16*)(w + off);       off += hBytes;
    _Float16* wtq = (_Float16*)(w + off);       off += wBytes;
    _Float16* wtk = (_Float16*)(w + off);       off += wBytes;
    _Float16* wtv = (_Float16*)(w + off);       off += wBytes;
    _Float16* wtp = (_Float16*)(w + off);       off += wBytes;
    _Float16* Qb  = (_Float16*)(w + off);       off += hBytes;
    _Float16* Kb  = (_Float16*)(w + off);       off += hBytes;
    _Float16* Vtb = (_Float16*)(w + off);       off += hBytes;
    _Float16* Ob  = (_Float16*)(w + off);       off += hBytes;
    (void)ws_size; (void)in_sizes; (void)n_in; (void)out_size;

    // 1) GroupNorm stats: 2 batches * 32 groups
    gn_stats_kernel<<<BATCH * NGRP, 256, 0, stream>>>(x, stats);
    // 2) normalize + f16 convert: B*N*C/4 threads
    gn_apply_kernel<<<(BATCH * NC / 4) / 256, 256, 0, stream>>>(x, gamma, beta, stats, hN);
    // 3) pack the four weight matrices (transpose + f16)
    pack_w_kernel<<<(CCH * CCH) / 256, 256, 0, stream>>>(wq, wtq);
    pack_w_kernel<<<(CCH * CCH) / 256, 256, 0, stream>>>(wk, wtk);
    pack_w_kernel<<<(CCH * CCH) / 256, 256, 0, stream>>>(wv, wtv);
    pack_w_kernel<<<(CCH * CCH) / 256, 256, 0, stream>>>(wp, wtp);
    // 4) Q, K (row-major), V (transposed) projections
    dim3 gg(NTOK / 128, CCH / 64, BATCH);
    qkv_gemm_kernel<<<gg, 256, 0, stream>>>(hN, wtq, bq, Qb, 0);
    qkv_gemm_kernel<<<gg, 256, 0, stream>>>(hN, wtk, bk, Kb, 0);
    qkv_gemm_kernel<<<gg, 256, 0, stream>>>(hN, wtv, bv, Vtb, 1);
    // 5) fused attention (async double-buffered K tiles)
    flash_attn_kernel<<<dim3(NTOK / 64, BATCH), 256, 0, stream>>>(Qb, Kb, Vtb, Ob);
    // 6) projection + bias + residual
    proj_gemm_kernel<<<gg, 256, 0, stream>>>(Ob, wtp, bp, x, out);
}
